// Renderer_fast_90065464197138
// MI455X (gfx1250) — matrix operand order repaired and checked by reading, not compiled, not run on hardware
//
#include <hip/hip_runtime.h>

// ============================================================================
// MI455X / gfx1250 renderer:
//   pass0: zero segment accumulators (L2-resident, 5.6 MB)
//   pass1: atomic segment reductions over 5.12M (id, z, dir, cos) entries
//   pass2: per-point features + 2-layer MLP on v_wmma_f32_16x16x32_f16
//          (16 points per wave; f16 operands, f32 accumulate)
//   pass3: gather feat[id] -> (1, 64, 800, 800), unit-stride stores per plane
// ============================================================================

typedef __attribute__((ext_vector_type(16))) _Float16 v16h;
typedef __attribute__((ext_vector_type(8)))  float    v8f;

constexpr int kH    = 800;
constexpr int kW    = 800;
constexpr int kP    = 8;
constexpr int kNPTS = 200000;
constexpr int kDIM  = 8;
constexpr int kHID  = 256;
constexpr int kN    = kH * kW * kP;   // 5,120,000 entries
constexpr int kPIX  = kH * kW;        // 640,000 pixels
constexpr float kZT = 0.2f;

constexpr int WAVES         = 4;              // waves per block (wave32)
constexpr int THREADS2      = WAVES * 32;     // 128
constexpr int PTS_PER_BLOCK = WAVES * 16;     // 64 points per block

// ---------------------------------------------------------------------------
// pass 0: init accumulators
// ---------------------------------------------------------------------------
__global__ void render_pass0_init(int* __restrict__ cnt, int* __restrict__ fidx,
                                  int* __restrict__ valid,
                                  float* __restrict__ sx, float* __restrict__ sy,
                                  float* __restrict__ sz, float* __restrict__ sc)
{
    const int i = blockIdx.x * blockDim.x + threadIdx.x;
    if (i >= kNPTS) return;
    cnt[i]   = 0;
    fidx[i]  = 0x7FFFFFFF;
    valid[i] = 0;
    sx[i] = 0.f; sy[i] = 0.f; sz[i] = 0.f; sc[i] = 0.f;
}

// ---------------------------------------------------------------------------
// pass 1: segment reductions via atomics (accumulators fit in L2 easily)
// ---------------------------------------------------------------------------
__global__ void render_pass1_reduce(const float* __restrict__ zbufs,
                                    const float* __restrict__ ray,
                                    const int* __restrict__ idbufs,
                                    int* __restrict__ cnt, int* __restrict__ fidx,
                                    int* __restrict__ valid,
                                    float* __restrict__ sx, float* __restrict__ sy,
                                    float* __restrict__ sz, float* __restrict__ sc)
{
    const int i = blockIdx.x * blockDim.x + threadIdx.x;
    if (i >= kN) return;
    const int   id  = idbufs[i];
    const float z   = zbufs[i];
    const int   pix = i >> 3;                       // / kP
    const float* r  = ray + (size_t)pix * 7;
    const float dx = r[3], dy = r[4], dz = r[5], cs = r[6];

    atomicAdd(&cnt[id], 1);
    atomicAdd(&sx[id], dx);
    atomicAdd(&sy[id], dy);
    atomicAdd(&sz[id], dz);
    atomicAdd(&sc[id], cs);
    atomicMin(&fidx[id], i);
    if (z > kZT) atomicOr(&valid[id], 1);
}

// ---------------------------------------------------------------------------
// pass 2: per-point feature build + MLP on WMMA
// One wave handles 16 points (one 16-row A tile).
// ---------------------------------------------------------------------------
__global__ __launch_bounds__(THREADS2)
void render_pass2_mlp(const float* __restrict__ zbufs, const float* __restrict__ ray,
                      const float* __restrict__ W1, const float* __restrict__ b1,
                      const float* __restrict__ W2, const float* __restrict__ b2,
                      const int* __restrict__ cnt, const int* __restrict__ fidx,
                      const int* __restrict__ valid,
                      const float* __restrict__ sx, const float* __restrict__ sy,
                      const float* __restrict__ sz, const float* __restrict__ sc,
                      float* __restrict__ feat)
{
    __shared__ _Float16 w1s[6][kHID];            // 3 KB
    __shared__ _Float16 w2s[kHID][kDIM];         // 4 KB
    __shared__ float    b1s[kHID];               // 1 KB
    __shared__ float    b2s[kDIM];
    __shared__ _Float16 xs[WAVES][16][8];        // per-point 6 inputs (padded to 8)
    __shared__ float    vms[WAVES][16];          // validity mask
    __shared__ _Float16 hid[WAVES][16][kHID + 8];// hidden, row pad -> no bank conflicts

    const int t = threadIdx.x;
    for (int k = t; k < 6 * kHID;    k += THREADS2) w1s[k / kHID][k % kHID] = (_Float16)W1[k];
    for (int k = t; k < kHID * kDIM; k += THREADS2) w2s[k / kDIM][k % kDIM] = (_Float16)W2[k];
    for (int k = t; k < kHID;        k += THREADS2) b1s[k] = b1[k];
    if (t < kDIM) b2s[t] = b2[t];

    const int wave = t >> 5;
    const int lane = t & 31;
    const int base = blockIdx.x * PTS_PER_BLOCK + wave * 16;

    // ---- per-point scalar features (lanes 0..15 = one point each) ----
    if (lane < 16) {
        const int m  = lane;
        const int pt = base + m;
        float x0 = 0.f, x1 = 0.f, x2 = 0.f, x3 = 0.f, x4 = 0.f, x5 = 0.f, vm = 0.f;
        if (pt < kNPTS) {
            const int c = cnt[pt];
            if (c > 0) {
                const float inv = 1.f / (float)c;       // safe = max(count,1) == count here
                const float dx = sx[pt] * inv;
                const float dy = sy[pt] * inv;
                const float dz = sz[pt] * inv;
                const float cm = sc[pt] * inv;          // cos >= 0.5 -> cm > 0
                int fi = fidx[pt];
                if (fi > kN - 1) fi = kN - 1;
                const float zf = zbufs[fi];
                const float s  = zf / cm;
                x0 = ray[0] + s * dx;                   // o is ray[0,0,0:3]
                x1 = ray[1] + s * dy;
                x2 = ray[2] + s * dz;
                x3 = dx; x4 = dy; x5 = dz;
                vm = (valid[pt] != 0) ? 1.f : 0.f;
            }
        }
        xs[wave][m][0] = (_Float16)x0; xs[wave][m][1] = (_Float16)x1;
        xs[wave][m][2] = (_Float16)x2; xs[wave][m][3] = (_Float16)x3;
        xs[wave][m][4] = (_Float16)x4; xs[wave][m][5] = (_Float16)x5;
        xs[wave][m][6] = (_Float16)0.f; xs[wave][m][7] = (_Float16)0.f;
        vms[wave][m] = vm;
    }
    __syncthreads();

    const int  nl     = lane & 15;
    const bool hiHalf = (lane >= 16);

    // ---- layer 1: A = x tile 16x32 (K=0..5 live), B = W1 column-tile 32x16 ----
    // A layout: lanes 0-15 halves[0..7]=K0..7, halves[8..15]=K16..23 (all pad->0);
    //           lanes 16-31 cover K8..15 / 24..31 (all 0).
    v16h a1;
    #pragma unroll
    for (int j = 0; j < 16; ++j) a1[j] = (_Float16)0.f;
    if (!hiHalf) {
        #pragma unroll
        for (int j = 0; j < 8; ++j) a1[j] = xs[wave][nl][j];
    }

    for (int tile = 0; tile < 16; ++tile) {
        const int ncol = tile * 16 + nl;
        // B layout: lanes 0-15 = col ncol, halves[0..15]=K0..15 (K>=6 zero);
        //           lanes 16-31 = K16..31 (all zero).
        v16h bf;
        #pragma unroll
        for (int j = 0; j < 16; ++j) bf[j] = (_Float16)0.f;
        if (!hiHalf) {
            #pragma unroll
            for (int j = 0; j < 6; ++j) bf[j] = w1s[j][ncol];
        }
        v8f cacc = {};
        cacc = __builtin_amdgcn_wmma_f32_16x16x32_f16(false, a1, false, bf,
                                                      (short)0, cacc, false, false);
        // fused bias + ReLU, stage hidden as f16
        const float bias = b1s[ncol];
        const int   mb   = hiHalf ? 8 : 0;
        #pragma unroll
        for (int i = 0; i < 8; ++i) {
            float v = cacc[i] + bias;
            v = v > 0.f ? v : 0.f;
            hid[wave][i + mb][ncol] = (_Float16)v;
        }
    }
    __syncthreads();

    // ---- layer 2: hidden(16x256) @ W2(256x8), 8 accumulating WMMAs ----
    v8f c2 = {};
    for (int f = 0; f < 8; ++f) {
        const int kb   = f * 32;
        const int aoff = kb + (hiHalf ? 8 : 0);
        v16h a, bf;
        #pragma unroll
        for (int j = 0; j < 8; ++j) a[j]     = hid[wave][nl][aoff + j];
        #pragma unroll
        for (int j = 0; j < 8; ++j) a[8 + j] = hid[wave][nl][aoff + 16 + j];

        const int kbb = kb + (hiHalf ? 8 : 0);
        if (nl < kDIM) {
            #pragma unroll
            for (int j = 0; j < 16; ++j) bf[j] = w2s[kbb + j + (j >= 8 ? 8 : 0)][nl];
        } else {
            #pragma unroll
            for (int j = 0; j < 16; ++j) bf[j] = (_Float16)0.f;  // pad cols 8..15
        }
        c2 = __builtin_amdgcn_wmma_f32_16x16x32_f16(false, a, false, bf,
                                                    (short)0, c2, false, false);
    }

    // ---- +b2, validity mask, store feature table ----
    if (nl < kDIM) {
        const int   mb   = hiHalf ? 8 : 0;
        const float bias = b2s[nl];
        #pragma unroll
        for (int i = 0; i < 8; ++i) {
            const int mm = i + mb;
            const int pt = base + mm;
            if (pt < kNPTS) {
                feat[pt * kDIM + nl] = (c2[i] + bias) * vms[wave][mm];
            }
        }
    }
}

// ---------------------------------------------------------------------------
// pass 3: gather feat[id] -> out[1, 64, H, W]; one thread per pixel.
// feat (6.4 MB) is L2-resident; each of 64 store streams is unit-stride in w.
// ---------------------------------------------------------------------------
__global__ void render_pass3_scatter(const int* __restrict__ idbufs,
                                     const float* __restrict__ feat,
                                     float* __restrict__ out)
{
    const int i = blockIdx.x * blockDim.x + threadIdx.x;   // pixel index
    if (i >= kPIX) return;
    #pragma unroll
    for (int p = 0; p < kP; ++p) {
        const int id = idbufs[i * kP + p];
        const float4* fp = (const float4*)(feat + (size_t)id * kDIM);
        const float4 f0 = fp[0];
        const float4 f1 = fp[1];
        float* o = out + (size_t)(p * kDIM) * kPIX + i;
        o[0 * (size_t)kPIX] = f0.x;
        o[1 * (size_t)kPIX] = f0.y;
        o[2 * (size_t)kPIX] = f0.z;
        o[3 * (size_t)kPIX] = f0.w;
        o[4 * (size_t)kPIX] = f1.x;
        o[5 * (size_t)kPIX] = f1.y;
        o[6 * (size_t)kPIX] = f1.z;
        o[7 * (size_t)kPIX] = f1.w;
    }
}

// ---------------------------------------------------------------------------
extern "C" void kernel_launch(void* const* d_in, const int* in_sizes, int n_in,
                              void* d_out, int out_size, void* d_ws, size_t ws_size,
                              hipStream_t stream)
{
    const float* zbufs = (const float*)d_in[0];
    const float* ray   = (const float*)d_in[1];
    const float* W1    = (const float*)d_in[2];
    const float* b1    = (const float*)d_in[3];
    const float* W2    = (const float*)d_in[4];
    const float* b2    = (const float*)d_in[5];
    const int*   idb   = (const int*)  d_in[6];
    float* out = (float*)d_out;

    // workspace layout (~12 MB): feat first (32B-aligned for float4 gathers)
    char* ws = (char*)d_ws;
    float* feat  = (float*)ws; ws += (size_t)kNPTS * kDIM * sizeof(float);
    int*   cnt   = (int*)ws;   ws += (size_t)kNPTS * sizeof(int);
    int*   fidx  = (int*)ws;   ws += (size_t)kNPTS * sizeof(int);
    int*   valid = (int*)ws;   ws += (size_t)kNPTS * sizeof(int);
    float* sx    = (float*)ws; ws += (size_t)kNPTS * sizeof(float);
    float* sy    = (float*)ws; ws += (size_t)kNPTS * sizeof(float);
    float* sz    = (float*)ws; ws += (size_t)kNPTS * sizeof(float);
    float* sc    = (float*)ws; ws += (size_t)kNPTS * sizeof(float);

    render_pass0_init<<<(kNPTS + 255) / 256, 256, 0, stream>>>(
        cnt, fidx, valid, sx, sy, sz, sc);

    render_pass1_reduce<<<(kN + 255) / 256, 256, 0, stream>>>(
        zbufs, ray, idb, cnt, fidx, valid, sx, sy, sz, sc);

    render_pass2_mlp<<<(kNPTS + PTS_PER_BLOCK - 1) / PTS_PER_BLOCK, THREADS2, 0, stream>>>(
        zbufs, ray, W1, b1, W2, b2, cnt, fidx, valid, sx, sy, sz, sc, feat);

    render_pass3_scatter<<<(kPIX + 255) / 256, 256, 0, stream>>>(idb, feat, out);
}
